// RelativeMultiHeadAttentionModule_3350074490969
// MI455X (gfx1250) — matrix operand order repaired and checked by reading, not compiled, not run on hardware
//
#include <hip/hip_runtime.h>
#include <hip/hip_bf16.h>

// ---------------------------------------------------------------------------
// Types for CDNA5 WMMA (wave32): v_wmma_f32_16x16x32_bf16
// ---------------------------------------------------------------------------
typedef __attribute__((ext_vector_type(16))) __bf16 v16bf;
typedef __attribute__((ext_vector_type(8)))  __bf16 v8bf;
typedef __attribute__((ext_vector_type(8)))  float  v8f;

union ABfrag { v16bf v; v8bf h[2]; };

__device__ __forceinline__ float  bf2f(__bf16 b) { return (float)b; }
__device__ __forceinline__ __bf16 f2bf(float f)  { return (__bf16)f; }

__device__ __forceinline__ v8f wmma_bf16(v16bf a, v16bf b, v8f c) {
  // (neg_a, A, neg_b, B, c_mod, C, reuse_a, reuse_b)
  return __builtin_amdgcn_wmma_f32_16x16x32_bf16(false, a, false, b, (short)0, c,
                                                 false, false);
}

// A fragment 16x32 bf16, row-major source with leading dim `ld` (elements).
// Lane L (<16): row, K 0..15 ; lane L+16: same row, K 16..31.
// `row` must already include (lane & 15).
__device__ __forceinline__ v16bf load_a_g(const __bf16* base, int ld, int row,
                                          int k0, int lane) {
  const __bf16* p = base + (size_t)row * ld + k0 + ((lane >> 4) & 1) * 16;
  ABfrag r;
  r.h[0] = *(const v8bf*)(p);
  r.h[1] = *(const v8bf*)(p + 8);
  return r.v;
}

// B fragment 32x16 bf16. Column n of B = row (n0+n) of source, contiguous
// along K. Lane n (<16): K 0..15 ; lane n+16: K 16..31.
__device__ __forceinline__ v16bf load_b_g(const __bf16* base, int ld, int n0,
                                          int k0, int lane) {
  const __bf16* p =
      base + (size_t)(n0 + (lane & 15)) * ld + k0 + ((lane >> 4) & 1) * 16;
  return *(const v16bf*)(p);
}

// ---------------------------------------------------------------------------
// CDNA5 async-to-LDS path (ASYNCcnt-tracked).  Inline asm for toolchain
// portability: the clang builtin arity differs between ROCm 7.2 and upstream.
// Low 32 bits of a generic LDS pointer == LDS byte offset (aperture layout).
// ---------------------------------------------------------------------------
__device__ __forceinline__ void async_copy_b128(const void* lds_dst,
                                                const void* gsrc) {
  unsigned int       loff = (unsigned int)(unsigned long long)lds_dst;
  unsigned long long ga   = (unsigned long long)gsrc;
  asm volatile("global_load_async_to_lds_b128 %0, %1, off"
               :
               : "v"(loff), "v"(ga)
               : "memory");
}

template <int N>
__device__ __forceinline__ void wait_asynccnt() {
  asm volatile("s_wait_asynccnt %0" : : "i"(N) : "memory");
}

// ---------------------------------------------------------------------------
// Constants
// ---------------------------------------------------------------------------
#define Dm 512
#define Hh 8
#define DH 64
#define Bb 8
#define Ss 1024
#define LN_EPS 6.1e-05f
#define RSCALE 0.04419417382415922f  // 1/sqrt(512)

enum { MODE_PLAIN = 0, MODE_QUQV = 1, MODE_VT = 2, MODE_OUTF = 3 };

// ---------------------------------------------------------------------------
// LayerNorm: x[row,512] -> bf16 xn.  One block per row.
// ---------------------------------------------------------------------------
__global__ __launch_bounds__(256) void k_ln(const float* __restrict__ x,
                                            const float* __restrict__ gamma,
                                            const float* __restrict__ beta,
                                            __bf16* __restrict__ xn) {
  __shared__ float red[256];
  const int row = blockIdx.x, tid = threadIdx.x;
  const float* xr = x + (size_t)row * Dm;
  float a = xr[tid], b = xr[tid + 256];
  red[tid] = a + b;
  __syncthreads();
  for (int s = 128; s > 0; s >>= 1) {
    if (tid < s) red[tid] += red[tid + s];
    __syncthreads();
  }
  const float mu = red[0] * (1.0f / Dm);
  __syncthreads();
  const float d0 = a - mu, d1 = b - mu;
  red[tid] = d0 * d0 + d1 * d1;
  __syncthreads();
  for (int s = 128; s > 0; s >>= 1) {
    if (tid < s) red[tid] += red[tid + s];
    __syncthreads();
  }
  const float rstd = rsqrtf(red[0] * (1.0f / Dm) + LN_EPS);
  xn[(size_t)row * Dm + tid]       = f2bf(d0 * rstd * gamma[tid] + beta[tid]);
  xn[(size_t)row * Dm + tid + 256] = f2bf(d1 * rstd * gamma[tid + 256] + beta[tid + 256]);
}

// ---------------------------------------------------------------------------
// f32 -> bf16 convert
// ---------------------------------------------------------------------------
__global__ __launch_bounds__(256) void k_cvt(const float* __restrict__ s,
                                             __bf16* __restrict__ d, int n) {
  int i = blockIdx.x * 256 + threadIdx.x;
  if (i < n) d[i] = f2bf(s[i]);
}

// ---------------------------------------------------------------------------
// Zero fill (uint4)
// ---------------------------------------------------------------------------
__global__ __launch_bounds__(256) void k_zero(uint4* __restrict__ p, size_t n) {
  size_t i = (size_t)blockIdx.x * 256 + threadIdx.x;
  if (i < n) p[i] = make_uint4(0u, 0u, 0u, 0u);
}

// ---------------------------------------------------------------------------
// Generic GEMM: C[M,N] = A[M,K] @ W[N,K]^T   (both operands K-contiguous)
// Block = 8 waves -> 64(M) x 128(N) tile; wave = 32x32 (4 WMMA C fragments).
// Epilogue selects output layout / bias / (u,v) add.
// ---------------------------------------------------------------------------
__global__ __launch_bounds__(256) void k_gemm(
    const __bf16* __restrict__ A, const __bf16* __restrict__ W, int M, int N,
    int K, int mode, const float* __restrict__ bias,
    const float* __restrict__ uvec, const float* __restrict__ vvec,
    __bf16* __restrict__ out0, __bf16* __restrict__ out1,
    float* __restrict__ outf) {
  const int lane = threadIdx.x & 31;
  const int wv = threadIdx.x >> 5;
  const int wm = wv >> 2, wn = wv & 3;
  const int m0 = blockIdx.x * 64 + wm * 32;
  const int n0 = blockIdx.y * 128 + wn * 32;

  v8f acc[2][2] = {};
  for (int k0 = 0; k0 < K; k0 += 32) {
    v16bf a0 = load_a_g(A, K, m0 + (lane & 15), k0, lane);
    v16bf a1 = load_a_g(A, K, m0 + 16 + (lane & 15), k0, lane);
    v16bf b0 = load_b_g(W, K, n0, k0, lane);
    v16bf b1 = load_b_g(W, K, n0 + 16, k0, lane);
    acc[0][0] = wmma_bf16(a0, b0, acc[0][0]);
    acc[0][1] = wmma_bf16(a0, b1, acc[0][1]);
    acc[1][0] = wmma_bf16(a1, b0, acc[1][0]);
    acc[1][1] = wmma_bf16(a1, b1, acc[1][1]);
  }

  const int cl = lane & 15, rh = (lane >> 4) * 8;
#pragma unroll
  for (int mi = 0; mi < 2; mi++)
#pragma unroll
    for (int ni = 0; ni < 2; ni++)
#pragma unroll
      for (int r = 0; r < 8; r++) {
        const int row = m0 + mi * 16 + r + rh;
        const int col = n0 + ni * 16 + cl;
        float val = acc[mi][ni][r];
        if (bias) val += bias[col];
        if (mode == MODE_PLAIN) {
          out0[(size_t)row * N + col] = f2bf(val);
        } else if (mode == MODE_QUQV) {
          out0[(size_t)row * N + col] = f2bf(val + uvec[col]);  // q+u
          out1[(size_t)row * N + col] = f2bf(val + vvec[col]);  // q+v
        } else if (mode == MODE_VT) {
          // store transposed per head: vT[b, h, dh, s]
          const int b = row >> 10, s = row & 1023;
          const int h = col >> 6, dh = col & 63;
          out0[((((size_t)b * Hh + h) * DH + dh) << 10) + s] = f2bf(val);
        } else {  // MODE_OUTF
          outf[(size_t)row * N + col] = val;
        }
      }
}

// ---------------------------------------------------------------------------
// BD GEMM with fused rel_shift scatter:
//   raw[i,r] = (q+v)[b,i,h,:] . p[r,h,:]
//   shifted flat index n = i*(S+1) + r + 1 - S  (drop if out of [0,S*S))
// ---------------------------------------------------------------------------
__global__ __launch_bounds__(256) void k_bd(const __bf16* __restrict__ qv,
                                            const __bf16* __restrict__ pb,
                                            __bf16* __restrict__ bd) {
  const int bh = blockIdx.z;
  const int b = bh >> 3, h = bh & 7;
  const int lane = threadIdx.x & 31;
  const int wv = threadIdx.x >> 5;
  const int wm = wv >> 2, wn = wv & 3;
  const int i0 = blockIdx.x * 64 + wm * 32;
  const int r0 = blockIdx.y * 128 + wn * 32;

  const __bf16* Ab = qv + ((size_t)b * Ss) * Dm + h * DH;
  const __bf16* Bp = pb + h * DH;

  v8f acc[2][2] = {};
#pragma unroll
  for (int k0 = 0; k0 < DH; k0 += 32) {
    v16bf a0 = load_a_g(Ab, Dm, i0 + (lane & 15), k0, lane);
    v16bf a1 = load_a_g(Ab, Dm, i0 + 16 + (lane & 15), k0, lane);
    v16bf b0 = load_b_g(Bp, Dm, r0, k0, lane);
    v16bf b1 = load_b_g(Bp, Dm, r0 + 16, k0, lane);
    acc[0][0] = wmma_bf16(a0, b0, acc[0][0]);
    acc[0][1] = wmma_bf16(a0, b1, acc[0][1]);
    acc[1][0] = wmma_bf16(a1, b0, acc[1][0]);
    acc[1][1] = wmma_bf16(a1, b1, acc[1][1]);
  }

  __bf16* slab = bd + (size_t)bh * Ss * Ss;
  const int cl = lane & 15, rh = (lane >> 4) * 8;
#pragma unroll
  for (int mi = 0; mi < 2; mi++)
#pragma unroll
    for (int ni = 0; ni < 2; ni++)
#pragma unroll
      for (int r = 0; r < 8; r++) {
        const long i = i0 + mi * 16 + r + rh;
        const long rr = r0 + ni * 16 + cl;
        const long n = i * (Ss + 1) + rr + 1 - Ss;  // rel_shift scatter
        if (n >= 0 && n < (long)Ss * Ss) slab[n] = f2bf(acc[mi][ni][r]);
      }
}

// ---------------------------------------------------------------------------
// Flash attention per (b,h).  Block = 8 waves x 16 q-rows = 128 q-rows.
// K/V tiles (32 keys) are staged cooperatively into LDS with CDNA5 async
// copies (GLOBAL_LOAD_ASYNC_TO_LDS_B128, ASYNCcnt), double-buffered: issue
// tile i+1, s_wait_asynccnt 2 (tile i done - async loads complete in order),
// barrier, compute tile i from LDS via WMMA, barrier.
// ---------------------------------------------------------------------------
__global__ __launch_bounds__(256) void k_attn(const __bf16* __restrict__ qu,
                                              const __bf16* __restrict__ kb,
                                              const __bf16* __restrict__ bd,
                                              const __bf16* __restrict__ vT,
                                              __bf16* __restrict__ ctx) {
  const int tid = threadIdx.x;
  const int lane = tid & 31;
  const int w = tid >> 5;
  const int bh = blockIdx.y;
  const int b = bh >> 3, h = bh & 7;
  const int m0 = blockIdx.x * 128 + w * 16;

  __shared__ __bf16 s_k[2][32][64];  // K tile: 32 keys x 64 dh (ld=64)
  __shared__ __bf16 s_v[2][64][32];  // V tile: 64 dh x 32 keys (ld=32)
  __shared__ float  s_sc[8][16][32];
  __shared__ __bf16 s_p[8][16][32];
  __shared__ float  s_m[8][16], s_l[8][16], s_f[8][16];

  const __bf16* qbase = qu + ((size_t)b * Ss) * Dm + h * DH;
  const __bf16* kbase = kb + ((size_t)b * Ss) * Dm + h * DH;
  const __bf16* vbase = vT + ((size_t)bh * DH) * Ss;
  const __bf16* bdsl = bd + (size_t)bh * Ss * Ss;

  // Cooperative async stage of one 32-key K/V tile (4KB + 4KB, 16B/thread).
  auto stage = [&](int buf, int t0) {
    {
      const int row = tid >> 3, seg = tid & 7;  // 32 rows x 8 x 16B
      async_copy_b128(&s_k[buf][row][seg * 8],
                      kbase + (size_t)(t0 + row) * Dm + seg * 8);
    }
    {
      const int row = tid >> 2, seg = tid & 3;  // 64 rows x 4 x 16B
      async_copy_b128(&s_v[buf][row][seg * 8],
                      vbase + (size_t)row * Ss + t0 + seg * 8);
    }
  };

  // Q fragments (16 rows x K=64) held for the whole pass
  const v16bf au0 = load_a_g(qbase, Dm, m0 + (lane & 15), 0, lane);
  const v16bf au1 = load_a_g(qbase, Dm, m0 + (lane & 15), 32, lane);

  v8f o[4] = {};
  if (lane < 16) {
    s_m[w][lane] = -1e30f;
    s_l[w][lane] = 0.0f;
  }

  stage(0, 0);  // prime the pipeline

  const int cl = lane & 15, rh = (lane >> 4) * 8;

  for (int it = 0; it < Ss / 32; ++it) {
    const int t0 = it * 32;
    const int cur = it & 1;

    if (it < Ss / 32 - 1) {
      stage(cur ^ 1, t0 + 32);  // prefetch next tile into other buffer
      wait_asynccnt<2>();       // current tile (2 older ops/wave) complete
    } else {
      wait_asynccnt<0>();
    }
    __syncthreads();  // staged data visible to all waves

    // ---- scores: AC (WMMA from LDS K tile) + pre-shifted BD, scaled ----
#pragma unroll
    for (int tt = 0; tt < 2; tt++) {
      const int t = t0 + tt * 16;
      v16bf bk0 = load_b_g(&s_k[cur][0][0], 64, tt * 16, 0, lane);
      v16bf bk1 = load_b_g(&s_k[cur][0][0], 64, tt * 16, 32, lane);
      v8f sc = {};
      sc = wmma_bf16(au0, bk0, sc);
      sc = wmma_bf16(au1, bk1, sc);
#pragma unroll
      for (int r = 0; r < 8; r++) {
        const int rl = r + rh;
        const float bdv = bf2f(bdsl[(size_t)(m0 + rl) * Ss + t + cl]);
        s_sc[w][rl][tt * 16 + cl] = (sc[r] + bdv) * RSCALE;
      }
    }
    __builtin_amdgcn_wave_barrier();

    // ---- online softmax over this 32-column chunk: lane i owns row i ----
    if (lane < 16) {
      const float mold = s_m[w][lane];
      float cm = -1e30f;
#pragma unroll
      for (int j = 0; j < 32; j++) cm = fmaxf(cm, s_sc[w][lane][j]);
      const float mnew = fmaxf(mold, cm);
      const float fac = __expf(mold - mnew);
      float sum = s_l[w][lane] * fac;
#pragma unroll
      for (int j = 0; j < 32; j++) {
        const float pe = __expf(s_sc[w][lane][j] - mnew);
        sum += pe;
        s_p[w][lane][j] = f2bf(pe);
      }
      s_m[w][lane] = mnew;
      s_l[w][lane] = sum;
      s_f[w][lane] = fac;
    }
    __builtin_amdgcn_wave_barrier();

    // ---- rescale O and accumulate P @ V from LDS (K=32 keys, N=64) ----
    const v16bf pa = load_a_g(&s_p[w][0][0], 32, lane & 15, 0, lane);
#pragma unroll
    for (int n = 0; n < 4; n++) {
#pragma unroll
      for (int r = 0; r < 8; r++) o[n][r] *= s_f[w][r + rh];
      v16bf vb = load_b_g(&s_v[cur][0][0], 32, n * 16, 0, lane);
      o[n] = wmma_bf16(pa, vb, o[n]);
    }
    __syncthreads();  // done with this buffer before it is restaged
  }

  // Normalize and write ctx[b, s, h*DH + dh] as bf16
#pragma unroll
  for (int n = 0; n < 4; n++)
#pragma unroll
    for (int r = 0; r < 8; r++) {
      const int rl = r + rh;
      const float val = o[n][r] / s_l[w][rl];
      ctx[((size_t)b * Ss + m0 + rl) * Dm + h * DH + n * 16 + cl] = f2bf(val);
    }
}

// ---------------------------------------------------------------------------
// Host launcher
// ---------------------------------------------------------------------------
extern "C" void kernel_launch(void* const* d_in, const int* in_sizes, int n_in,
                              void* d_out, int out_size, void* d_ws,
                              size_t ws_size, hipStream_t stream) {
  (void)in_sizes; (void)n_in; (void)out_size; (void)ws_size;

  const float* x     = (const float*)d_in[0];
  // d_in[1] = mask (all false) -- unused
  const float* pos   = (const float*)d_in[2];
  const float* Wq    = (const float*)d_in[3];
  const float* bq    = (const float*)d_in[4];
  const float* Wk    = (const float*)d_in[5];
  const float* bk    = (const float*)d_in[6];
  const float* Wv    = (const float*)d_in[7];
  const float* bv    = (const float*)d_in[8];
  const float* Wpos  = (const float*)d_in[9];
  const float* Wout  = (const float*)d_in[10];
  const float* bout  = (const float*)d_in[11];
  const float* uu    = (const float*)d_in[12];
  const float* vv    = (const float*)d_in[13];
  const float* gamma = (const float*)d_in[14];
  const float* beta  = (const float*)d_in[15];

  char* ws = (char*)d_ws;
  size_t off = 0;
  auto alloc = [&](size_t bytes) -> char* {
    char* p = ws + off;
    off += (bytes + 255) & ~(size_t)255;
    return p;
  };

  const size_t BS = (size_t)Bb * Ss;              // 8192 rows
  __bf16* xnb  = (__bf16*)alloc(BS * Dm * 2);     // LayerNorm output
  __bf16* posb = (__bf16*)alloc((size_t)Ss * Dm * 2);
  __bf16* wqb  = (__bf16*)alloc((size_t)Dm * Dm * 2);
  __bf16* wkb  = (__bf16*)alloc((size_t)Dm * Dm * 2);
  __bf16* wvb  = (__bf16*)alloc((size_t)Dm * Dm * 2);
  __bf16* wpb  = (__bf16*)alloc((size_t)Dm * Dm * 2);
  __bf16* wob  = (__bf16*)alloc((size_t)Dm * Dm * 2);
  __bf16* qub  = (__bf16*)alloc(BS * Dm * 2);     // q + u
  __bf16* qvb  = (__bf16*)alloc(BS * Dm * 2);     // q + v
  __bf16* kbb  = (__bf16*)alloc(BS * Dm * 2);     // k
  __bf16* vtb  = (__bf16*)alloc(BS * Dm * 2);     // v transposed [b,h,dh,s]
  __bf16* pbb  = (__bf16*)alloc((size_t)Ss * Dm * 2);  // p projection
  __bf16* bdb  = (__bf16*)alloc((size_t)Bb * Hh * Ss * Ss * 2);  // shifted BD
  __bf16* ctxb = (__bf16*)alloc(BS * Dm * 2);     // attention output

  // 1) LayerNorm
  k_ln<<<(int)BS, 256, 0, stream>>>(x, gamma, beta, xnb);

  // 2) Converts (weights + pos_emb)
  k_cvt<<<1024, 256, 0, stream>>>(Wq, wqb, Dm * Dm);
  k_cvt<<<1024, 256, 0, stream>>>(Wk, wkb, Dm * Dm);
  k_cvt<<<1024, 256, 0, stream>>>(Wv, wvb, Dm * Dm);
  k_cvt<<<1024, 256, 0, stream>>>(Wpos, wpb, Dm * Dm);
  k_cvt<<<1024, 256, 0, stream>>>(Wout, wob, Dm * Dm);
  k_cvt<<<2048, 256, 0, stream>>>(pos, posb, Ss * Dm);

  // 3) Projections (WMMA GEMMs)
  dim3 gP(128, 4);
  k_gemm<<<gP, 256, 0, stream>>>(xnb, wqb, (int)BS, Dm, Dm, MODE_QUQV, bq, uu,
                                 vv, qub, qvb, nullptr);
  k_gemm<<<gP, 256, 0, stream>>>(xnb, wkb, (int)BS, Dm, Dm, MODE_PLAIN, bk,
                                 nullptr, nullptr, kbb, nullptr, nullptr);
  k_gemm<<<gP, 256, 0, stream>>>(xnb, wvb, (int)BS, Dm, Dm, MODE_VT, bv,
                                 nullptr, nullptr, vtb, nullptr, nullptr);
  dim3 gp2(16, 4);
  k_gemm<<<gp2, 256, 0, stream>>>(posb, wpb, Ss, Dm, Dm, MODE_PLAIN, nullptr,
                                  nullptr, nullptr, pbb, nullptr, nullptr);

  // 4) Zero BD slab (rel_shift leaves some slots untouched)
  const size_t bd16 = ((size_t)Bb * Hh * Ss * Ss * 2) / 16;  // uint4 count
  k_zero<<<(unsigned)((bd16 + 255) / 256), 256, 0, stream>>>((uint4*)bdb, bd16);

  // 5) BD GEMM with fused rel_shift scatter
  dim3 gBD(16, 8, 64);
  k_bd<<<gBD, 256, 0, stream>>>(qvb, pbb, bdb);

  // 6) Flash attention (async double-buffered K/V staging)
  dim3 gA(8, 64);
  k_attn<<<gA, 256, 0, stream>>>(qub, kbb, bdb, vtb, ctxb);

  // 7) Output projection -> f32 d_out
  k_gemm<<<gP, 256, 0, stream>>>(ctxb, wob, (int)BS, Dm, Dm, MODE_OUTF, bout,
                                 nullptr, nullptr, nullptr, nullptr,
                                 (float*)d_out);
}
